// TestSAN_10582799417884
// MI455X (gfx1250) — matrix operand order repaired and checked by reading, not compiled, not run on hardware
//
#include <hip/hip_runtime.h>
#include <hip/hip_bf16.h>

typedef __attribute__((ext_vector_type(16))) __bf16 v16bf;
typedef __attribute__((ext_vector_type(8)))  float  v8f;
typedef __attribute__((ext_vector_type(4)))  float  v4f;

#define FDIM 64
#define ODIM 64

struct ProjArgs {
    const float* W[3];
    float*       h[3];
    const float* as[3];
    const float* ad[3];
    float*       sv[3];
    float*       dv[3];
};

// Fused projection kernel: one read of x produces up to 3 h = x@W matrices
// (bf16 WMMA, fp32 accumulate) plus per-GAT attention dots s = h@a_s, d = h@a_d.
// Block = 128 threads = 4 waves; each wave computes a 16x64 output tile.
template<int NW>
__global__ __launch_bounds__(128)
void proj_kernel(const float* __restrict__ x, int n, ProjArgs p) {
    // Weights staged in LDS, pre-swizzled into the exact WMMA B-fragment layout:
    // [wi][kstep(2)][ntile(4)][lane(32)][e(16)]  (16 bf16 = 32B contiguous per lane)
    __shared__ v16bf lds_w[NW * 256];
    __bf16* lds_b = (__bf16*)lds_w;

    const int tid = threadIdx.x;
    for (int idx = tid; idx < NW * 4096; idx += 128) {
        int wi   = idx >> 12;
        int rem  = idx & 4095;
        int ks   = rem >> 11;
        int nt   = (rem >> 9) & 3;
        int lane = (rem >> 4) & 31;
        int e    = rem & 15;
        // B-fragment mapping: K = ks*32 + (lane/16)*8 + (e&7) + (e>>3)*16; N = nt*16 + lane%16
        int k = ks * 32 + (lane >> 4) * 8 + (e & 7) + (e >> 3) * 16;
        int nn = nt * 16 + (lane & 15);
        lds_b[idx] = (__bf16)p.W[wi][k * ODIM + nn];
    }
    __syncthreads();

    const int wave   = tid >> 5;
    const int lane   = tid & 31;
    const int lane16 = lane & 15;
    const int hi     = lane >> 4;
    const int tile   = blockIdx.x * 4 + wave;
    const int row_base = tile * 16;
    if (row_base >= n) return;   // n is a multiple of 16; only whole-tile exit

    v8f acc[NW][4];
    #pragma unroll
    for (int wi = 0; wi < NW; ++wi)
        #pragma unroll
        for (int nt = 0; nt < 4; ++nt)
            #pragma unroll
            for (int j = 0; j < 8; ++j) acc[wi][nt][j] = 0.0f;

    const float* xr = x + (size_t)(row_base + lane16) * FDIM;

    #pragma unroll
    for (int ks = 0; ks < 2; ++ks) {
        // A-fragment (16-bit 16x32 layout): lane holds row m=lane%16,
        // K = ks*32 + hi*8 + [0..7] and ks*32 + 16 + hi*8 + [0..7]
        const int kb = ks * 32 + hi * 8;
        v4f c0 = *(const v4f*)(xr + kb);
        v4f c1 = *(const v4f*)(xr + kb + 4);
        v4f c2 = *(const v4f*)(xr + kb + 16);
        v4f c3 = *(const v4f*)(xr + kb + 20);
        v16bf a;
        #pragma unroll
        for (int j = 0; j < 4; ++j) {
            a[j]      = (__bf16)c0[j];
            a[4 + j]  = (__bf16)c1[j];
            a[8 + j]  = (__bf16)c2[j];
            a[12 + j] = (__bf16)c3[j];
        }
        #pragma unroll
        for (int wi = 0; wi < NW; ++wi) {
            #pragma unroll
            for (int nt = 0; nt < 4; ++nt) {
                const v16bf b = lds_w[((wi * 2 + ks) * 4 + nt) * 32 + lane];
                acc[wi][nt] = __builtin_amdgcn_wmma_f32_16x16x32_bf16(
                    false, a, false, b, (short)0, acc[wi][nt], false, false);
            }
        }
    }

    // Store h tiles; fused attention dot-products for GAT weights.
    #pragma unroll
    for (int wi = 0; wi < NW; ++wi) {
        float* hw = p.h[wi];
        #pragma unroll
        for (int nt = 0; nt < 4; ++nt)
            #pragma unroll
            for (int r = 0; r < 8; ++r)
                hw[(size_t)(row_base + r + hi * 8) * ODIM + nt * 16 + lane16] = acc[wi][nt][r];

        if (p.as[wi]) {
            float ps[8], pd[8];
            #pragma unroll
            for (int r = 0; r < 8; ++r) { ps[r] = 0.f; pd[r] = 0.f; }
            #pragma unroll
            for (int nt = 0; nt < 4; ++nt) {
                const float avs = p.as[wi][nt * 16 + lane16];
                const float avd = p.ad[wi][nt * 16 + lane16];
                #pragma unroll
                for (int r = 0; r < 8; ++r) {
                    ps[r] += acc[wi][nt][r] * avs;
                    pd[r] += acc[wi][nt][r] * avd;
                }
            }
            // reduce across the 16 lanes sharing a row (wave32, width-16 xor shuffle)
            #pragma unroll
            for (int off = 8; off > 0; off >>= 1) {
                #pragma unroll
                for (int r = 0; r < 8; ++r) {
                    ps[r] += __shfl_xor(ps[r], off, 16);
                    pd[r] += __shfl_xor(pd[r], off, 16);
                }
            }
            if (lane16 == 0) {
                #pragma unroll
                for (int r = 0; r < 8; ++r) {
                    p.sv[wi][row_base + r + hi * 8] = ps[r];
                    p.dv[wi][row_base + r + hi * 8] = pd[r];
                }
            }
        }
    }
}

// Order-preserving encoding so float segment-max can use u32 atomicMax.
// enc(f) is monotone; enc value 0 corresponds to below -FLT_MAX (== -inf init via memset 0).
__device__ __forceinline__ unsigned enc_f32(float f) {
    unsigned b = __float_as_uint(f);
    return (b & 0x80000000u) ? ~b : (b | 0x80000000u);
}
__device__ __forceinline__ float dec_f32(unsigned u) {
    unsigned b = (u & 0x80000000u) ? (u & 0x7FFFFFFFu) : ~u;
    return __uint_as_float(b);
}

__global__ void edge_logits(const int* __restrict__ rows, const int* __restrict__ cols,
                            const float* __restrict__ s, const float* __restrict__ d,
                            float* __restrict__ e_buf, unsigned* __restrict__ m_u, int E) {
    int t = blockIdx.x * blockDim.x + threadIdx.x;
    if (t >= E) return;
    int r = rows[t];
    float e = s[r] + d[cols[t]];
    e = e > 0.f ? e : 0.2f * e;            // leaky_relu(0.2)
    e_buf[t] = e;
    atomicMax(m_u + r, enc_f32(e));
}

__global__ void edge_exp(const int* __restrict__ rows, float* __restrict__ e_buf,
                         const unsigned* __restrict__ m_u, float* __restrict__ den, int E) {
    int t = blockIdx.x * blockDim.x + threadIdx.x;
    if (t >= E) return;
    int r = rows[t];
    float ex = __expf(e_buf[t] - dec_f32(m_u[r]));
    e_buf[t] = ex;
    atomicAdd(den + r, ex);
}

// One thread per (edge, column). den==nullptr -> weighted SpMM path (coef = vals),
// else GAT path (coef = ex / den[row]).
__global__ void scatter_kernel(const int* __restrict__ rows, const int* __restrict__ cols,
                               const float* __restrict__ coef, const float* __restrict__ den,
                               const float* __restrict__ h, float* __restrict__ out,
                               long long total) {
    long long t = (long long)blockIdx.x * blockDim.x + threadIdx.x;
    if (t >= total) return;
    int edge = (int)(t >> 6);
    int c    = (int)(t & 63);
    int r  = rows[edge];
    int cc = cols[edge];
    float a = coef[edge];
    if (den) a /= den[r];
    atomicAdd(out + (size_t)r * ODIM + c, a * h[(size_t)cc * ODIM + c]);
}

__global__ void relu_kernel(float* __restrict__ o, long long n) {
    long long t = (long long)blockIdx.x * blockDim.x + threadIdx.x;
    if (t < n) o[t] = fmaxf(o[t], 0.0f);
}

extern "C" void kernel_launch(void* const* d_in, const int* in_sizes, int n_in,
                              void* d_out, int out_size, void* d_ws, size_t ws_size,
                              hipStream_t stream) {
    (void)in_sizes; (void)n_in; (void)ws_size;
    constexpr int N0 = 50000, N1 = 150000, N2 = 100000;
    constexpr int E0 = 400000, E1 = 1200000, E2 = 800000;

    const float* x0    = (const float*)d_in[0];
    const float* x1    = (const float*)d_in[1];
    const float* x2    = (const float*)d_in[2];
    const float* w1p   = (const float*)d_in[3];
    const float* w1d   = (const float*)d_in[4];
    const float* a1d_s = (const float*)d_in[5];
    const float* a1d_d = (const float*)d_in[6];
    const float* w2p   = (const float*)d_in[7];
    const float* w2u   = (const float*)d_in[8];
    const float* a2u_s = (const float*)d_in[9];
    const float* a2u_d = (const float*)d_in[10];
    const float* w2d   = (const float*)d_in[11];
    const float* a2d_s = (const float*)d_in[12];
    const float* a2d_d = (const float*)d_in[13];
    const float* w3p   = (const float*)d_in[14];
    const float* w3u   = (const float*)d_in[15];
    const float* a3u_s = (const float*)d_in[16];
    const float* a3u_d = (const float*)d_in[17];
    const int* l0_rows  = (const int*)d_in[18];
    const int* l0_cols  = (const int*)d_in[19];
    const float* l0_vals = (const float*)d_in[20];
    const int* l1u_rows = (const int*)d_in[21];
    const int* l1u_cols = (const int*)d_in[22];
    const int* l1d_rows = (const int*)d_in[23];
    const int* l1d_cols = (const int*)d_in[24];
    const int* l1_rows  = (const int*)d_in[25];
    const int* l1_cols  = (const int*)d_in[26];
    const float* l1_vals = (const float*)d_in[27];
    const int* l2_rows  = (const int*)d_in[28];
    const int* l2_cols  = (const int*)d_in[29];
    const float* l2_vals = (const float*)d_in[30];

    float* out0 = (float*)d_out;
    float* out1 = out0 + (size_t)N0 * ODIM;
    float* out2 = out1 + (size_t)N1 * ODIM;

    // Workspace carve-out (regions reused across layers; stream-ordered).
    char* wsb = (char*)d_ws;
    size_t off = 0;
    auto take = [&](size_t bytes) -> char* {
        char* ptr = wsb + off;
        off += (bytes + 255) & ~(size_t)255;
        return ptr;
    };
    float*    hA    = (float*)take((size_t)N1 * ODIM * 4);
    float*    hB    = (float*)take((size_t)N1 * ODIM * 4);
    float*    hC    = (float*)take((size_t)N1 * ODIM * 4);
    float*    e_buf = (float*)take((size_t)E1 * 4);
    unsigned* m_u   = (unsigned*)take((size_t)N1 * 4);
    float*    den   = (float*)take((size_t)N1 * 4);
    float*    svA   = (float*)take((size_t)N1 * 4);
    float*    dvA   = (float*)take((size_t)N1 * 4);
    float*    svB   = (float*)take((size_t)N1 * 4);
    float*    dvB   = (float*)take((size_t)N1 * 4);

    hipMemsetAsync(d_out, 0, (size_t)out_size * sizeof(float), stream);

    auto run_gat = [&](const int* rows, const int* cols, const float* sv, const float* dv,
                       const float* h, float* out, int Nn, int Ee) {
        hipMemsetAsync(m_u, 0, (size_t)Nn * 4, stream);   // 0 encodes -inf
        hipMemsetAsync(den, 0, (size_t)Nn * 4, stream);
        edge_logits<<<(Ee + 255) / 256, 256, 0, stream>>>(rows, cols, sv, dv, e_buf, m_u, Ee);
        edge_exp<<<(Ee + 255) / 256, 256, 0, stream>>>(rows, e_buf, m_u, den, Ee);
        long long tot = (long long)Ee * ODIM;
        scatter_kernel<<<(int)((tot + 255) / 256), 256, 0, stream>>>(rows, cols, e_buf, den, h, out, tot);
    };
    auto run_spmm = [&](const int* rows, const int* cols, const float* vals,
                        const float* h, float* out, int Ee) {
        long long tot = (long long)Ee * ODIM;
        scatter_kernel<<<(int)((tot + 255) / 256), 256, 0, stream>>>(rows, cols, vals, nullptr, h, out, tot);
    };

    // ---- layer 1: GAT(w1d) + spmm(w1p) on x0 / L0 ----
    {
        ProjArgs p = {};
        p.W[0] = w1d; p.h[0] = hA; p.as[0] = a1d_s; p.ad[0] = a1d_d; p.sv[0] = svA; p.dv[0] = dvA;
        p.W[1] = w1p; p.h[1] = hB;
        proj_kernel<2><<<(N0 / 16 + 3) / 4, 128, 0, stream>>>(x0, N0, p);
        run_gat(l0_rows, l0_cols, svA, dvA, hA, out0, N0, E0);
        run_spmm(l0_rows, l0_cols, l0_vals, hB, out0, E0);
    }
    // ---- layer 2: GAT(w2u) + GAT(w2d) + spmm(w2p) on x1 / L1 ----
    {
        ProjArgs p = {};
        p.W[0] = w2u; p.h[0] = hA; p.as[0] = a2u_s; p.ad[0] = a2u_d; p.sv[0] = svA; p.dv[0] = dvA;
        p.W[1] = w2d; p.h[1] = hB; p.as[1] = a2d_s; p.ad[1] = a2d_d; p.sv[1] = svB; p.dv[1] = dvB;
        p.W[2] = w2p; p.h[2] = hC;
        proj_kernel<3><<<(N1 / 16 + 3) / 4, 128, 0, stream>>>(x1, N1, p);
        run_gat(l1u_rows, l1u_cols, svA, dvA, hA, out1, N1, E1);
        run_gat(l1d_rows, l1d_cols, svB, dvB, hB, out1, N1, E1);
        run_spmm(l1_rows, l1_cols, l1_vals, hC, out1, E1);
    }
    // ---- layer 3: GAT(w3u) + spmm(w3p) on x2 / L2 ----
    {
        ProjArgs p = {};
        p.W[0] = w3u; p.h[0] = hA; p.as[0] = a3u_s; p.ad[0] = a3u_d; p.sv[0] = svA; p.dv[0] = dvA;
        p.W[1] = w3p; p.h[1] = hB;
        proj_kernel<2><<<(N2 / 16 + 3) / 4, 128, 0, stream>>>(x2, N2, p);
        run_gat(l2_rows, l2_cols, svA, dvA, hA, out2, N2, E2);
        run_spmm(l2_rows, l2_cols, l2_vals, hB, out2, E2);
    }

    long long nout = (long long)out_size;
    relu_kernel<<<(int)((nout + 255) / 256), 256, 0, stream>>>((float*)d_out, nout);
}
